// ONLinear_6751688589355
// MI455X (gfx1250) — compile-verified, matrix operand order checked by reading, not run on hardware
//
#include <hip/hip_runtime.h>
#include <stdint.h>

// ---------------------------------------------------------------------------
// ONLinear: block-quantized weight dequant + GEMM (y = x @ W^T + b)
//   M = 8192, N = 4096, K = 4096  -> 274.9 GFLOP, compute-bound on MI455X.
// Pipeline:
//   1) dequant W -> bf16 in d_ws                      (32 MiB)
//   2) convert x fp32 -> bf16 in d_ws                 (64 MiB)
//   3) bf16 WMMA GEMM, LDS double-buffered via GLOBAL_LOAD_ASYNC_TO_LDS_B128
//      (zero-VGPR staging, ASYNCcnt-tracked), v_wmma_f32_16x16x32_bf16.
// ---------------------------------------------------------------------------

#define OUTF 4096
#define INF  4096
#define MROWS 8192
#define NUM_BLOCKS (OUTF * INF / 16)

#define BM 128
#define BN 128
#define BK 32
#define LDT 40                   // padded LDS row stride (bf16): 32 + 8
#define BUFB (BM * LDT * 2)      // LDS bytes per buffer = 10240
#define ROWB (LDT * 2)           // LDS bytes per row    = 80

typedef __attribute__((ext_vector_type(16))) __bf16 v16bf;
typedef __attribute__((ext_vector_type(8)))  float  v8f;

union FragU { v16bf v; uint4 q[2]; };

// Async DMA: 16 bytes global -> LDS, no VGPR data path (ASYNCcnt). [CDNA5]
__device__ __forceinline__ void async_ld16(uint32_t lds_off, uint64_t gaddr) {
    asm volatile("global_load_async_to_lds_b128 %0, %1, off"
                 :: "v"(lds_off), "v"(gaddr) : "memory");
}
__device__ __forceinline__ void wait_async0() {
    asm volatile("s_wait_asynccnt 0x0" ::: "memory");
}

// Pack two fp32 -> two bf16 (round-half-up) in one v_perm_b32.
__device__ __forceinline__ uint32_t pkbf2(float a, float b) {
    uint32_t ua = __builtin_bit_cast(uint32_t, a) + 0x8000u;
    uint32_t ub = __builtin_bit_cast(uint32_t, b) + 0x8000u;
    return __builtin_amdgcn_perm(ub, ua, 0x07060302u);
}

// ---------------------------------------------------------------------------
__global__ void __launch_bounds__(256) dequant_w_kernel(
    const float* __restrict__ cent, const int* __restrict__ labels,
    const float* __restrict__ bins, const int* __restrict__ nidx,
    uint16_t* __restrict__ wq)
{
    int b = blockIdx.x * 256 + threadIdx.x;
    float s = bins[nidx[b]];
    const float4* c = (const float4*)(cent + (size_t)labels[b] * 16);
    float4 c0 = c[0], c1 = c[1], c2 = c[2], c3 = c[3];
    uint4 lo, hi;
    lo.x = pkbf2(c0.x * s, c0.y * s); lo.y = pkbf2(c0.z * s, c0.w * s);
    lo.z = pkbf2(c1.x * s, c1.y * s); lo.w = pkbf2(c1.z * s, c1.w * s);
    hi.x = pkbf2(c2.x * s, c2.y * s); hi.y = pkbf2(c2.z * s, c2.w * s);
    hi.z = pkbf2(c3.x * s, c3.y * s); hi.w = pkbf2(c3.z * s, c3.w * s);
    uint4* dst = (uint4*)(wq + (size_t)b * 16);
    dst[0] = lo; dst[1] = hi;
}

__global__ void __launch_bounds__(256) cvt_x_kernel(
    const float* __restrict__ X, uint16_t* __restrict__ Xb)
{
    size_t t = (size_t)blockIdx.x * 256 + threadIdx.x;
    const float4* src = (const float4*)(X + t * 8);
    float4 f0 = src[0], f1 = src[1];
    uint4 o;
    o.x = pkbf2(f0.x, f0.y); o.y = pkbf2(f0.z, f0.w);
    o.z = pkbf2(f1.x, f1.y); o.w = pkbf2(f1.z, f1.w);
    *(uint4*)(Xb + t * 8) = o;
}

// ---------------------------------------------------------------------------
// GEMM: grid (N/BN, M/BM) = (32, 64), 256 threads = 8 waves (4M x 2N),
// each wave owns 2x4 tiles of 16x16 (32x64 region).
// ---------------------------------------------------------------------------
template <int APRE>
__global__ void __launch_bounds__(256, 1) gemm_bf16_kernel(
    const float*    __restrict__ Xf,
    const uint16_t* __restrict__ Xb,
    const uint16_t* __restrict__ Wq,
    const float*    __restrict__ bias,
    float*          __restrict__ Y)
{
    __shared__ uint16_t sA[2][BM][LDT];
    __shared__ uint16_t sB[2][BN][LDT];

    const int tid  = threadIdx.x;
    const int lane = tid & 31;
    const int wid  = tid >> 5;
    const int m0 = blockIdx.y * BM;
    const int n0 = blockIdx.x * BN;

    const int wm = (wid & 3) * 32;
    const int wn = (wid >> 2) * 64;

    // DMA mapping: 128x32 bf16 tile -> 2 x 16B per thread
    const int br = tid >> 2;            // 0..63 ; rows br, br+64
    const int bc = (tid & 3) * 8;
    // fp32 A staging mapping (APRE=0): 4 float4 per thread
    const int ar = tid >> 3;
    const int ac = (tid & 7) * 4;

    // Global source addresses (advanced in place each stage)
    uint64_t gB0 = (uint64_t)(uintptr_t)(Wq + (size_t)(n0 + br) * INF + bc);
    uint64_t gB1 = gB0 + (uint64_t)64 * INF * 2;
    uint64_t gA0 = 0, gA1 = 0;
    if (APRE) {
        gA0 = (uint64_t)(uintptr_t)(Xb + (size_t)(m0 + br) * INF + bc);
        gA1 = gA0 + (uint64_t)64 * INF * 2;
    }
    // LDS DMA destinations (buffer 0 bases; buffer offset added per stage)
    const uint32_t ldsB0 = (uint32_t)(uintptr_t)&sB[0][br][bc];
    const uint32_t ldsA0 = (uint32_t)(uintptr_t)&sA[0][br][bc];

    // LDS read bases for compute: single VGPR each, all else immediate offsets
    const int r = lane & 15;
    const int h = (lane >> 4) * 8;      // K-half select per ISA A/B layout
    const char* aP = (const char*)&sA[0][wm + r][h];
    const char* bP = (const char*)&sB[0][wn + r][h];

    v8f acc[2][4] = {};
    float4 raf[4];                      // APRE=0 only

    auto issueAsync = [&](uint32_t bo) {
        async_ld16(ldsB0 + bo,             gB0);
        async_ld16(ldsB0 + bo + 64 * ROWB, gB1);
        if (APRE) {
            async_ld16(ldsA0 + bo,             gA0);
            async_ld16(ldsA0 + bo + 64 * ROWB, gA1);
        }
        gB0 += BK * 2; gB1 += BK * 2;
        if (APRE) { gA0 += BK * 2; gA1 += BK * 2; }
    };
    auto loadA = [&](int kt) {          // APRE=0
        const int k0 = kt * BK;
#pragma unroll
        for (int i = 0; i < 4; ++i)
            raf[i] = *(const float4*)(Xf + (size_t)(m0 + ar + i * 32) * INF + k0 + ac);
    };
    auto storeA = [&](uint32_t bo) {    // APRE=0
        uint16_t* base = (uint16_t*)((char*)&sA[0][0][0] + bo);
#pragma unroll
        for (int i = 0; i < 4; ++i) {
            uint2 p;
            p.x = pkbf2(raf[i].x, raf[i].y);
            p.y = pkbf2(raf[i].z, raf[i].w);
            *(uint2*)(base + (ar + i * 32) * LDT + ac) = p;
        }
    };

    auto compute = [&](uint32_t bo) {
        const char* aB = aP + bo;
        const char* bB = bP + bo;
        FragU a0, a1;
        a0.q[0] = *(const uint4*)(aB);
        a0.q[1] = *(const uint4*)(aB + 32);
        a1.q[0] = *(const uint4*)(aB + 16 * ROWB);
        a1.q[1] = *(const uint4*)(aB + 16 * ROWB + 32);
#pragma unroll
        for (int in = 0; in < 4; ++in) {
            FragU b;
            b.q[0] = *(const uint4*)(bB + in * 16 * ROWB);
            b.q[1] = *(const uint4*)(bB + in * 16 * ROWB + 32);
            acc[0][in] = __builtin_amdgcn_wmma_f32_16x16x32_bf16(
                false, a0.v, false, b.v, (short)0, acc[0][in], false, false);
            acc[1][in] = __builtin_amdgcn_wmma_f32_16x16x32_bf16(
                false, a1.v, false, b.v, (short)0, acc[1][in], false, false);
        }
    };

    // Prologue: DMA stage 0 into buffer 0
    issueAsync(0);
    if (!APRE) { loadA(0); storeA(0); }

    const int NK = INF / BK;            // 128 stages
    uint32_t bo = 0;                    // current buffer byte offset (0 / BUFB)
#pragma unroll 1
    for (int kt = 0; kt < NK; ++kt) {
        wait_async0();                  // drain DMA into current buffer
        __syncthreads();                // visible to all; prior readers done
        if (kt + 1 < NK) {
            issueAsync(BUFB - bo);      // DMA next tile into other buffer
            if (!APRE) loadA(kt + 1);
        }
        compute(bo);
        if (!APRE && kt + 1 < NK) storeA(BUFB - bo);
        bo = BUFB - bo;                 // toggle buffers (uniform SALU)
    }

    // Epilogue: C/D layout — lane holds N = lane%16; VGPR r holds M = r + 8*(lane/16)
    const int cn = lane & 15;
    const int ch = lane >> 4;
#pragma unroll
    for (int im = 0; im < 2; ++im) {
#pragma unroll
        for (int in = 0; in < 4; ++in) {
            const int ng  = n0 + wn + in * 16 + cn;
            const int mg0 = m0 + wm + im * 16 + ch * 8;
            const float bv = bias[ng];
#pragma unroll
            for (int rr = 0; rr < 8; ++rr)
                Y[(size_t)(mg0 + rr) * OUTF + ng] = acc[im][in][rr] + bv;
        }
    }
}

// ---------------------------------------------------------------------------
extern "C" void kernel_launch(void* const* d_in, const int* in_sizes, int n_in,
                              void* d_out, int out_size, void* d_ws, size_t ws_size,
                              hipStream_t stream) {
    const float* x      = (const float*)d_in[0];
    const float* cent   = (const float*)d_in[1];
    const int*   labels = (const int*)  d_in[2];
    const float* bins   = (const float*)d_in[3];
    const int*   nidx   = (const int*)  d_in[4];
    const float* bias   = (const float*)d_in[5];
    float*       y      = (float*)d_out;

    const size_t wbytes = (size_t)OUTF * INF * 2;      // 32 MiB
    const size_t xbytes = (size_t)MROWS * INF * 2;     // 64 MiB
    uint16_t* wq = (uint16_t*)d_ws;
    uint16_t* xb = (uint16_t*)((char*)d_ws + wbytes);

    dequant_w_kernel<<<NUM_BLOCKS / 256, 256, 0, stream>>>(cent, labels, bins, nidx, wq);

    dim3 grid(OUTF / BN, MROWS / BM);                  // (32, 64)
    if (ws_size >= wbytes + xbytes) {
        const size_t nx = (size_t)MROWS * INF / 8;
        cvt_x_kernel<<<(int)(nx / 256), 256, 0, stream>>>(x, xb);
        gemm_bf16_kernel<1><<<grid, 256, 0, stream>>>(x, xb, wq, bias, y);
    } else {
        gemm_bf16_kernel<0><<<grid, 256, 0, stream>>>(x, xb, wq, bias, y);
    }
}
